// HADM_17858474017085
// MI455X (gfx1250) — matrix operand order repaired and checked by reading, not compiled, not run on hardware
//
#include <hip/hip_runtime.h>
#include <hip/hip_bf16.h>
#include <stdint.h>

typedef __attribute__((ext_vector_type(16))) _Float16 v16h;
typedef __attribute__((ext_vector_type(8)))  float    v8f;

#define XSTRIDE 264   // padded row stride (f16 elems) for 81x256 LDS tile
#define BSS     72    // sizes FC B-tile in LDS; rows use unpadded stride 32
#define DW_MAXC 4.135166556742356f
#define IMGF    800.0f

__device__ __forceinline__ unsigned short f2h_u(float x) {
  _Float16 h = (_Float16)x;
  return __builtin_bit_cast(unsigned short, h);
}

union AF { v16h v; uint4 q[2]; };

// ---------------------------------------------------------------------------
// Conv-weight transpose: conv_w[s][l][oc][ic][ky][kx] (f32)
//   -> wtc[(s*4+l)][tap][oc][ic] (f16), tap = ky*3+kx
// ---------------------------------------------------------------------------
__global__ void wtrans_kernel(const float* __restrict__ cw,
                              unsigned short* __restrict__ wtc) {
  long idx = (long)blockIdx.x * 256 + threadIdx.x; // total 3*4*9*256*256
  if (idx >= 7077888L) return;
  int ic  = (int)(idx & 255);
  int oc  = (int)((idx >> 8) & 255);
  int tap = (int)((idx >> 16) % 9);
  int sl  = (int)(idx / 589824L);
  size_t in = ((size_t)(sl * 256 + oc) * 256 + ic) * 9 + tap;
  wtc[idx] = f2h_u(cw[in]);
}

// ---------------------------------------------------------------------------
// Head kernel: one workgroup per ROI. ROI-align -> 4x (conv3x3 WMMA + LN + relu)
// LDS holds the zero-padded 9x9x256 f16 activation tile the whole time.
// ---------------------------------------------------------------------------
__global__ __launch_bounds__(256) void head_kernel(
    int s,
    const float* __restrict__ p2, const float* __restrict__ p3,
    const float* __restrict__ p4, const float* __restrict__ p5,
    const float* __restrict__ boxes,           // [512][4]
    const unsigned short* __restrict__ wtc,    // [12][9][256][256] f16
    const float* __restrict__ lng, const float* __restrict__ lnb, // [3][4][256]
    unsigned short* __restrict__ feats)        // [512][12544] f16 (c*49+pos)
{
  __shared__ unsigned short xpad[81 * XSTRIDE];
  __shared__ float smean[64];
  __shared__ float svar[64];

  const int tid  = threadIdx.x;
  const int roi  = blockIdx.x;
  const int lane = tid & 31;
  const int wave = tid >> 5;

  for (int i = tid; i < 81 * XSTRIDE; i += 256) xpad[i] = 0;

  // ---- level assignment (all threads redundantly) ----
  float b0 = boxes[roi * 4 + 0], b1 = boxes[roi * 4 + 1];
  float b2 = boxes[roi * 4 + 2], b3 = boxes[roi * 4 + 3];
  float area = (b2 - b0) * (b3 - b1);
  float lv = floorf(4.f + log2f(sqrtf(fmaxf(area, 1.f)) / 224.f + 1e-8f));
  int lvl = (int)fminf(fmaxf(lv, 2.f), 5.f) - 2;
  const float* feat; int H; float scale;
  if      (lvl == 0) { feat = p2; H = 200; scale = 0.25f;    }
  else if (lvl == 1) { feat = p3; H = 100; scale = 0.125f;   }
  else if (lvl == 2) { feat = p4; H = 50;  scale = 0.0625f;  }
  else               { feat = p5; H = 25;  scale = 0.03125f; }
  const int W = H;
  float x1 = b0 * scale, y1 = b1 * scale, x2 = b2 * scale, y2 = b3 * scale;
  float bw = fmaxf(x2 - x1, 1.f) / 7.f;
  float bh = fmaxf(y2 - y1, 1.f) / 7.f;
  __syncthreads();

  // ---- ROI align: thread == channel ----
  {
    const int ic = tid;
    const float* fc_ = feat + (size_t)ic * H * W;
    for (int pos = 0; pos < 49; ++pos) {
      int py = pos / 7, px = pos % 7;
      float acc = 0.f;
#pragma unroll
      for (int sy = 0; sy < 2; ++sy)
#pragma unroll
        for (int sx = 0; sx < 2; ++sx) {
          float yy = y1 + (py + (sy + 0.5f) * 0.5f) * bh;
          float xx = x1 + (px + (sx + 0.5f) * 0.5f) * bw;
          bool valid = (yy > -1.f) && (yy < (float)H) && (xx > -1.f) && (xx < (float)W);
          float y = fminf(fmaxf(yy, 0.f), (float)(H - 1));
          float x = fminf(fmaxf(xx, 0.f), (float)(W - 1));
          float yf = floorf(y), xf = floorf(x);
          int y0 = (int)yf, x0 = (int)xf;
          int y1i = min(y0 + 1, H - 1), x1i = min(x0 + 1, W - 1);
          float ly = y - yf, lx = x - xf;
          float v00 = fc_[y0 * W + x0],  v01 = fc_[y0 * W + x1i];
          float v10 = fc_[y1i * W + x0], v11 = fc_[y1i * W + x1i];
          float v = v00 * (1 - ly) * (1 - lx) + v01 * (1 - ly) * lx +
                    v10 * ly * (1 - lx) + v11 * ly * lx;
          acc += valid ? v : 0.f;
        }
      xpad[((py + 1) * 9 + (px + 1)) * XSTRIDE + ic] = f2h_u(acc * 0.25f);
    }
  }
  __syncthreads();

  const int klo = (lane >> 4) << 3;   // 0 for lanes 0-15, 8 for lanes 16-31

  // ---- 4 conv layers ----
  for (int l = 0; l < 4; ++l) {
    if (tid < 64) { smean[tid] = 0.f; svar[tid] = 0.f; }
    __syncthreads();

    v8f acc[4][2];
#pragma unroll
    for (int mt = 0; mt < 4; ++mt)
#pragma unroll
      for (int nt = 0; nt < 2; ++nt)
#pragma unroll
        for (int v = 0; v < 8; ++v) acc[mt][nt][v] = 0.f;

    const unsigned short* wl = wtc + (size_t)((s * 4 + l) * 9) * 65536;

#pragma unroll 1
    for (int tap = 0; tap < 9; ++tap) {
      const int ty = tap / 3, tx = tap % 3;
      const unsigned short* wt = wl + (size_t)tap * 65536 + wave * 32 * 256;
#pragma unroll 1
      for (int icb = 0; icb < 8; ++icb) {
        AF a[4];
#pragma unroll
        for (int mt = 0; mt < 4; ++mt) {
          int m = min(mt * 16 + (lane & 15), 48);
          int srow = (m / 7 + ty) * 9 + (m % 7 + tx);
          int base = srow * XSTRIDE + icb * 32 + klo;
          a[mt].q[0] = *(const uint4*)&xpad[base];
          a[mt].q[1] = *(const uint4*)&xpad[base + 16];
        }
        const unsigned short* wp = wt + (size_t)(lane & 15) * 256 + icb * 32 + klo;
        AF bq0, bq1;
        bq0.q[0] = *(const uint4*)(wp);
        bq0.q[1] = *(const uint4*)(wp + 16);
        bq1.q[0] = *(const uint4*)(wp + 16 * 256);
        bq1.q[1] = *(const uint4*)(wp + 16 * 256 + 16);
#pragma unroll
        for (int mt = 0; mt < 4; ++mt) {
          acc[mt][0] = __builtin_amdgcn_wmma_f32_16x16x32_f16(
              false, a[mt].v, false, bq0.v, (short)0, acc[mt][0], false, false);
          acc[mt][1] = __builtin_amdgcn_wmma_f32_16x16x32_f16(
              false, a[mt].v, false, bq1.v, (short)0, acc[mt][1], false, false);
        }
      }
    }

    // ---- LayerNorm stats from registers: reduce 32 channels per row ----
#pragma unroll
    for (int mt = 0; mt < 4; ++mt)
#pragma unroll
      for (int v = 0; v < 8; ++v) {
        float e0 = acc[mt][0][v], e1 = acc[mt][1][v];
        float sum = e0 + e1, sq = e0 * e0 + e1 * e1;
        for (int m2 = 1; m2 < 16; m2 <<= 1) {
          sum += __shfl_xor(sum, m2, 32);
          sq  += __shfl_xor(sq,  m2, 32);
        }
        if ((lane & 15) == 0) {
          int row = mt * 16 + v + klo;
          atomicAdd(&smean[row], sum);
          atomicAdd(&svar[row],  sq);
        }
      }
    __syncthreads();

    // ---- normalize + relu, store back into padded tile ----
#pragma unroll
    for (int mt = 0; mt < 4; ++mt)
#pragma unroll
      for (int nt = 0; nt < 2; ++nt) {
        int col = wave * 32 + nt * 16 + (lane & 15);
        float g  = lng[(s * 4 + l) * 256 + col];
        float bb = lnb[(s * 4 + l) * 256 + col];
#pragma unroll
        for (int v = 0; v < 8; ++v) {
          int row = mt * 16 + v + klo;
          float mu  = smean[row] * (1.f / 256.f);
          float var = svar[row] * (1.f / 256.f) - mu * mu;
          float xv = (acc[mt][nt][v] - mu) * rsqrtf(var + 1e-6f) * g + bb;
          xv = fmaxf(xv, 0.f);
          if (row < 49) {
            int prow = (row / 7 + 1) * 9 + (row % 7) + 1;
            xpad[prow * XSTRIDE + col] = f2h_u(xv);
          }
        }
      }
    __syncthreads();
  }

  // ---- flatten to feats[roi][c*49+pos] ----
  for (int k = tid; k < 12544; k += 256) {
    int ic = k / 49, pos = k % 49;
    int prow = (pos / 7 + 1) * 9 + (pos % 7) + 1;
    feats[(size_t)roi * 12544 + k] = xpad[prow * XSTRIDE + ic];
  }
}

// ---------------------------------------------------------------------------
// FC: h = relu(feats(512x12544 f16) @ fw^T (1024x12544 f32) + fb), WMMA GEMM.
// Grid (8, 16): 64x64 output tile per workgroup, B staged/transposed via LDS.
// ---------------------------------------------------------------------------
__global__ __launch_bounds__(256) void fc_kernel(
    const unsigned short* __restrict__ feats,
    const float* __restrict__ fw, const float* __restrict__ fb,
    float* __restrict__ h)
{
  __shared__ unsigned short Bs[32 * BSS];
  const int tid = threadIdx.x, lane = tid & 31, wave = tid >> 5;
  const int mt = wave & 3, np = wave >> 2;
  const int bm = blockIdx.x, bn = blockIdx.y;

  v8f acc0, acc1;
#pragma unroll
  for (int v = 0; v < 8; ++v) { acc0[v] = 0.f; acc1[v] = 0.f; }

  const int klo = (lane >> 4) << 3;
  const int roi = bm * 64 + mt * 16 + (lane & 15);
  const unsigned short* arow = feats + (size_t)roi * 12544;

  const int nn = tid >> 2;           // 0..63 local fc column
  const int kk0 = (tid & 3) * 8;     // 0,8,16,24 local k
  const float* fwrow = fw + (size_t)(bn * 64 + nn) * 12544 + kk0;

  for (int k0 = 0; k0 < 12544; k0 += 32) {
    __builtin_prefetch(fwrow + k0 + 512, 0, 1);
    float4 f0 = *(const float4*)(fwrow + k0);
    float4 f1 = *(const float4*)(fwrow + k0 + 4);
    Bs[nn * 32 + kk0 + 0] = f2h_u(f0.x);
    Bs[nn * 32 + kk0 + 1] = f2h_u(f0.y);
    Bs[nn * 32 + kk0 + 2] = f2h_u(f0.z);
    Bs[nn * 32 + kk0 + 3] = f2h_u(f0.w);
    Bs[nn * 32 + kk0 + 4] = f2h_u(f1.x);
    Bs[nn * 32 + kk0 + 5] = f2h_u(f1.y);
    Bs[nn * 32 + kk0 + 6] = f2h_u(f1.z);
    Bs[nn * 32 + kk0 + 7] = f2h_u(f1.w);
    __syncthreads();

    AF a;
    a.q[0] = *(const uint4*)(arow + k0 + klo);
    a.q[1] = *(const uint4*)(arow + k0 + klo + 16);
    const int nb0 = np * 32;
    AF bq0, bq1;
    bq0.q[0] = *(const uint4*)&Bs[(nb0 + (lane & 15)) * 32 + klo];
    bq0.q[1] = *(const uint4*)&Bs[(nb0 + (lane & 15)) * 32 + klo + 16];
    bq1.q[0] = *(const uint4*)&Bs[(nb0 + 16 + (lane & 15)) * 32 + klo];
    bq1.q[1] = *(const uint4*)&Bs[(nb0 + 16 + (lane & 15)) * 32 + klo + 16];

    acc0 = __builtin_amdgcn_wmma_f32_16x16x32_f16(false, a.v, false, bq0.v,
                                                  (short)0, acc0, false, false);
    acc1 = __builtin_amdgcn_wmma_f32_16x16x32_f16(false, a.v, false, bq1.v,
                                                  (short)0, acc1, false, false);
    __syncthreads();
  }

#pragma unroll
  for (int v = 0; v < 8; ++v) {
    int row = bm * 64 + mt * 16 + v + klo;
    int c0 = bn * 64 + np * 32 + (lane & 15);
    int c1 = c0 + 16;
    h[(size_t)row * 1024 + c0] = fmaxf(acc0[v] + fb[c0], 0.f);
    h[(size_t)row * 1024 + c1] = fmaxf(acc1[v] + fb[c1], 0.f);
  }
}

// ---------------------------------------------------------------------------
// cls/box dots: logits[roi][0..6]=cls, [7..10]=box deltas
// ---------------------------------------------------------------------------
__global__ void logits_kernel(const float* __restrict__ h,
                              const float* __restrict__ cw, const float* __restrict__ cb,
                              const float* __restrict__ bw, const float* __restrict__ bb,
                              float* __restrict__ logits) {
  int id = blockIdx.x * blockDim.x + threadIdx.x;
  if (id >= 512 * 11) return;
  int roi = id / 11, j = id % 11;
  const float* w; float bias;
  if (j < 7) { w = cw + (size_t)j * 1024; bias = cb[j]; }
  else       { w = bw + (size_t)(j - 7) * 1024; bias = bb[j - 7]; }
  const float* hr = h + (size_t)roi * 1024;
  float s = 0.f;
  for (int k = 0; k < 1024; k += 4) {
    float4 a = *(const float4*)(hr + k);
    float4 b = *(const float4*)(w + k);
    s += a.x * b.x + a.y * b.y + a.z * b.z + a.w * b.w;
  }
  logits[roi * 11 + j] = s + bias;
}

// ---------------------------------------------------------------------------
// softmax + score accumulate + box decode/clip
// ---------------------------------------------------------------------------
__global__ void decode_kernel(const float* __restrict__ logits,
                              const float* __restrict__ boxesIn,
                              float* __restrict__ boxesOut,
                              float* __restrict__ scoresAcc,
                              int s, float wx, float wy, float ww, float wh) {
  int roi = blockIdx.x * blockDim.x + threadIdx.x;
  if (roi >= 512) return;
  const float* lg = logits + roi * 11;
  float m = lg[0];
  for (int c = 1; c < 7; ++c) m = fmaxf(m, lg[c]);
  float e[7]; float sum = 0.f;
  for (int c = 0; c < 7; ++c) { e[c] = expf(lg[c] - m); sum += e[c]; }
  for (int c = 0; c < 7; ++c) {
    float p = e[c] / sum;
    if (s == 0) scoresAcc[roi * 7 + c] = p; else scoresAcc[roi * 7 + c] += p;
  }
  float a0 = boxesIn[roi * 4 + 0], a1 = boxesIn[roi * 4 + 1];
  float a2 = boxesIn[roi * 4 + 2], a3 = boxesIn[roi * 4 + 3];
  float ax = (a0 + a2) * 0.5f, ay = (a1 + a3) * 0.5f;
  float aw = a2 - a0, ah = a3 - a1;
  float dx = lg[7] / wx, dy = lg[8] / wy;
  float dw = fminf(lg[9] / ww, DW_MAXC), dh = fminf(lg[10] / wh, DW_MAXC);
  float cx = dx * aw + ax, cy = dy * ah + ay;
  float bw2 = expf(dw) * aw, bh2 = expf(dh) * ah;
  boxesOut[roi * 4 + 0] = fminf(fmaxf(cx - bw2 * 0.5f, 0.f), IMGF);
  boxesOut[roi * 4 + 1] = fminf(fmaxf(cy - bh2 * 0.5f, 0.f), IMGF);
  boxesOut[roi * 4 + 2] = fminf(fmaxf(cx + bw2 * 0.5f, 0.f), IMGF);
  boxesOut[roi * 4 + 3] = fminf(fmaxf(cy + bh2 * 0.5f, 0.f), IMGF);
}

// ---------------------------------------------------------------------------
// Sequential NMS (100 dets) in one workgroup, all state LDS-resident.
// ---------------------------------------------------------------------------
__global__ __launch_bounds__(256) void nms_kernel(
    const float* __restrict__ scoresAcc, const float* __restrict__ boxes,
    float* __restrict__ out) {
  __shared__ float sc[3072];
  __shared__ float bx[512 * 4];
  __shared__ float rv[256];
  __shared__ int   ri[256];
  const int t = threadIdx.x;

  for (int c = t; c < 3072; c += 256) {
    int cls = c >> 9, roi = c & 511;
    float v = scoresAcc[roi * 7 + cls] * (1.f / 3.f);
    sc[c] = (v > 0.05f) ? v : -1.f;
  }
  for (int i = t; i < 2048; i += 256) bx[i] = boxes[i];
  __syncthreads();

  for (int it = 0; it < 100; ++it) {
    float best = -INFINITY; int bi = 0;
    for (int c = t; c < 3072; c += 256) {
      float v = sc[c];
      if (v > best) { best = v; bi = c; }
    }
    rv[t] = best; ri[t] = bi;
    __syncthreads();
    for (int off = 128; off > 0; off >>= 1) {
      if (t < off) {
        float vo = rv[t + off]; int io = ri[t + off];
        if (vo > rv[t] || (vo == rv[t] && io < ri[t])) { rv[t] = vo; ri[t] = io; }
      }
      __syncthreads();
    }
    int i = ri[0]; float si = rv[0];
    int li = i >> 9, ir = i & 511;
    float ib0 = bx[ir * 4 + 0], ib1 = bx[ir * 4 + 1];
    float ib2 = bx[ir * 4 + 2], ib3 = bx[ir * 4 + 3];
    if (t == 0) {
      bool valid = si > 0.05f;
      float vf = valid ? 1.f : 0.f;
      out[it * 4 + 0] = ib0 * vf; out[it * 4 + 1] = ib1 * vf;
      out[it * 4 + 2] = ib2 * vf; out[it * 4 + 3] = ib3 * vf;
      out[400 + it] = valid ? si : 0.f;
      out[500 + it] = valid ? (float)(li + 1) : 0.f;
    }
    float a1 = (ib2 - ib0) * (ib3 - ib1);
    for (int c = t; c < 3072; c += 256) {
      if ((c >> 9) == li) {
        int r = c & 511;
        float xx1 = fmaxf(ib0, bx[r * 4 + 0]);
        float yy1 = fmaxf(ib1, bx[r * 4 + 1]);
        float xx2 = fminf(ib2, bx[r * 4 + 2]);
        float yy2 = fminf(ib3, bx[r * 4 + 3]);
        float inter = fmaxf(xx2 - xx1, 0.f) * fmaxf(yy2 - yy1, 0.f);
        float a2 = (bx[r * 4 + 2] - bx[r * 4 + 0]) * (bx[r * 4 + 3] - bx[r * 4 + 1]);
        float iou = inter / (a1 + a2 - inter + 1e-9f);
        if (iou > 0.5f || c == i) sc[c] = -INFINITY;
      } else if (c == i) {
        sc[c] = -INFINITY;
      }
    }
    __syncthreads();
  }
}

// ---------------------------------------------------------------------------
extern "C" void kernel_launch(void* const* d_in, const int* in_sizes, int n_in,
                              void* d_out, int out_size, void* d_ws, size_t ws_size,
                              hipStream_t stream) {
  const float* p2        = (const float*)d_in[0];
  const float* p3        = (const float*)d_in[1];
  const float* p4        = (const float*)d_in[2];
  const float* p5        = (const float*)d_in[3];
  const float* proposals = (const float*)d_in[4];
  const float* conv_w    = (const float*)d_in[5];
  const float* ln_g      = (const float*)d_in[6];
  const float* ln_b      = (const float*)d_in[7];
  const float* fc_w      = (const float*)d_in[8];
  const float* fc_b      = (const float*)d_in[9];
  const float* cls_w     = (const float*)d_in[10];
  const float* cls_b     = (const float*)d_in[11];
  const float* box_w     = (const float*)d_in[12];
  const float* box_b     = (const float*)d_in[13];
  float* out = (float*)d_out;

  uint8_t* ws = (uint8_t*)d_ws;
  size_t o = 0;
  unsigned short* wtc    = (unsigned short*)(ws + o); o += 14155776;  // 12*9*256*256*2
  unsigned short* feats  = (unsigned short*)(ws + o); o += 12845056;  // 512*12544*2
  float* hbuf            = (float*)(ws + o);          o += 2097152;   // 512*1024*4
  float* logits          = (float*)(ws + o);          o += 22528;     // 512*11*4
  float* boxesb          = (float*)(ws + o);          o += 8192;      // 512*4*4
  float* scores          = (float*)(ws + o);          o += 14336;     // 512*7*4

  wtrans_kernel<<<27648, 256, 0, stream>>>(conv_w, wtc);

  const float CW[3][4] = {{10.f, 10.f, 5.f, 5.f},
                          {20.f, 20.f, 10.f, 10.f},
                          {30.f, 30.f, 15.f, 15.f}};
  for (int s = 0; s < 3; ++s) {
    const float* bin = (s == 0) ? proposals : boxesb;
    head_kernel<<<512, 256, 0, stream>>>(s, p2, p3, p4, p5, bin, wtc,
                                         ln_g, ln_b, feats);
    fc_kernel<<<dim3(8, 16), 256, 0, stream>>>(feats,
                                               fc_w + (size_t)s * 1024 * 12544,
                                               fc_b + s * 1024, hbuf);
    logits_kernel<<<22, 256, 0, stream>>>(hbuf,
                                          cls_w + (size_t)s * 7 * 1024, cls_b + s * 7,
                                          box_w + (size_t)s * 4 * 1024, box_b + s * 4,
                                          logits);
    decode_kernel<<<2, 256, 0, stream>>>(logits, bin, boxesb, scores, s,
                                         CW[s][0], CW[s][1], CW[s][2], CW[s][3]);
  }
  nms_kernel<<<1, 256, 0, stream>>>(scores, boxesb, out);
}
